// EndpointVectorField_11038065950782
// MI455X (gfx1250) — matrix operand order repaired and checked by reading, not compiled, not run on hardware
//
#include <hip/hip_runtime.h>

typedef __bf16 bf;
typedef __attribute__((ext_vector_type(16))) __bf16 v16bf;
typedef __attribute__((ext_vector_type(8)))  __bf16 v8bf;
typedef __attribute__((ext_vector_type(8)))  float  v8f;

#define E_TOTAL   1000000
#define N_TOTAL   100000
#define FDIM      64
#define INDIM     208
#define KPAD      224      // 7 * 32
#define NKPAD     96       // 3 * 32 (80 padded)
#define EDGE_TILES (E_TOTAL / 16)   // 62500
#define NODE_TILES (N_TOTAL / 16)   // 6250

#define LDS_ORDER() asm volatile("" ::: "memory")

__device__ __forceinline__ float sigm(float x) { return 1.f / (1.f + __expf(-x)); }
__device__ __forceinline__ float silu(float x) { return x * sigm(x); }

__device__ __forceinline__ v16bf cvt16(float4 f0, float4 f1, float4 f2, float4 f3) {
  v16bf r;
  r[0]=(bf)f0.x; r[1]=(bf)f0.y; r[2]=(bf)f0.z; r[3]=(bf)f0.w;
  r[4]=(bf)f1.x; r[5]=(bf)f1.y; r[6]=(bf)f1.z; r[7]=(bf)f1.w;
  r[8]=(bf)f2.x; r[9]=(bf)f2.y; r[10]=(bf)f2.z; r[11]=(bf)f2.w;
  r[12]=(bf)f3.x; r[13]=(bf)f3.y; r[14]=(bf)f3.z; r[15]=(bf)f3.w;
  return r;
}
__device__ __forceinline__ v8bf cvt8(float4 f0, float4 f1) {
  v8bf r;
  r[0]=(bf)f0.x; r[1]=(bf)f0.y; r[2]=(bf)f0.z; r[3]=(bf)f0.w;
  r[4]=(bf)f1.x; r[5]=(bf)f1.y; r[6]=(bf)f1.z; r[7]=(bf)f1.w;
  return r;
}

// A fragment (16x32 bf16) from global f32 row pointer. ISA layout:
// lane (row=l&15, half=l>>4, k0=half*8): elems j=p[k0+j], 8+j=p[16+k0+j]
__device__ __forceinline__ v16bf gmemA(const float* p, int k0) {
  const float4* a = reinterpret_cast<const float4*>(p + k0);
  const float4* b = reinterpret_cast<const float4*>(p + 16 + k0);
  return cvt16(a[0], a[1], b[0], b[1]);
}

// A fragment from LDS bf16 row-major [16][stride]
__device__ __forceinline__ v16bf ldsA(const bf* base, int row, int stride, int kb32, int k0) {
  const v8bf lo = *reinterpret_cast<const v8bf*>(base + row*stride + kb32 + k0);
  const v8bf hi = *reinterpret_cast<const v8bf*>(base + row*stride + kb32 + 16 + k0);
  v16bf r;
#pragma unroll
  for (int j = 0; j < 8; ++j) { r[j] = lo[j]; r[8+j] = hi[j]; }
  return r;
}

// 4 N-tiles of WMMA against transposed bf16 weight Wt[N][strideK] in LDS.
// B 32x16 layout: lane (col=l&15, half): elem j = Wt[col][kb32+half*8+j], elem 8+j = Wt[col][kb32+16+half*8+j]
__device__ __forceinline__ void wmma4(v8f* acc, v16bf a, const bf* Wt, int strideK,
                                      int kb32, int col, int half) {
#pragma unroll
  for (int n = 0; n < 4; ++n) {
    const bf* p = Wt + (col + 16*n) * strideK + kb32 + half*8;
    const v8bf lo = *reinterpret_cast<const v8bf*>(p);
    const v8bf hi = *reinterpret_cast<const v8bf*>(p + 16);
    v16bf bfr;
#pragma unroll
    for (int j = 0; j < 8; ++j) { bfr[j] = lo[j]; bfr[8+j] = hi[j]; }
    acc[n] = __builtin_amdgcn_wmma_f32_16x16x32_bf16(
        false, a, false, bfr, (short)0, acc[n], false, false);
  }
}

// ===================== Edge update kernel =====================
__global__ __launch_bounds__(128)
void edge_update_kernel(const float* __restrict__ ns, const float* __restrict__ ef,
                        const float* __restrict__ dd, const int* __restrict__ src,
                        const int* __restrict__ dst,
                        const float* __restrict__ W1, const float* __restrict__ b1,
                        const float* __restrict__ W2, const float* __restrict__ b2,
                        const float* __restrict__ lng, const float* __restrict__ lnb,
                        float* __restrict__ out) {
  __shared__ __align__(16) bf    sW1t[64 * KPAD];      // 28672 B, zero-padded K 208..223
  __shared__ __align__(16) bf    sW2t[64 * 64];        // 8192 B
  __shared__ __align__(16) bf    sH[4][16 * 64];       // per-wave layer-1 output staging
  __shared__ __align__(16) float sEF[4][16 * 64];      // per-wave edge_feats staging

  for (int i = threadIdx.x; i < 64 * KPAD; i += blockDim.x) {
    int n = i / KPAD, k = i % KPAD;
    sW1t[i] = (bf)((k < INDIM) ? W1[k * FDIM + n] : 0.f);
  }
  for (int i = threadIdx.x; i < 64 * 64; i += blockDim.x) {
    int n = i >> 6, k = i & 63;
    sW2t[i] = (bf)W2[k * FDIM + n];
  }
  __syncthreads();

  const int lane = threadIdx.x & 31;
  const int w    = threadIdx.x >> 5;
  const int col  = lane & 15;
  const int half = lane >> 4;
  const int k0   = half * 8;

  float bias1[4], bias2[4], gam[4], bet[4];
#pragma unroll
  for (int n = 0; n < 4; ++n) {
    int c = col + 16*n;
    bias1[n] = b1[c]; bias2[n] = b2[c]; gam[n] = lng[c]; bet[n] = lnb[c];
  }

  bf*    hstage  = sH[w];
  float* efstage = sEF[w];

  const int t0      = blockIdx.x * 4 + w;
  const int tstride = gridDim.x * 4;

  // software-pipelined gather indices (one tile ahead)
  int is_n = 0, id_n = 0;
  if (t0 < EDGE_TILES) { is_n = src[t0*16 + col]; id_n = dst[t0*16 + col]; }

  for (int t = t0; t < EDGE_TILES; t += tstride) {
    const int ebase = t * 16;
    const int erow  = ebase + col;
    const int is = is_n, id = id_n;
    const int tn = t + tstride;
    if (tn < EDGE_TILES) {          // issue next-tile index loads early
      is_n = src[tn*16 + col];
      id_n = dst[tn*16 + col];
    }
    const float* rs = ns + (size_t)is * 64;
    const float* rd = ns + (size_t)id * 64;
    const float* re = ef + (size_t)erow * 64;
    const float* rr = dd + (size_t)erow * 16;

    v8f acc[4];
#pragma unroll
    for (int n = 0; n < 4; ++n) { v8f v;
#pragma unroll
      for (int r = 0; r < 8; ++r) v[r] = bias1[n]; acc[n] = v; }

    // K blocks 0..3: src / dst node scalars
    wmma4(acc, gmemA(rs,      k0), sW1t, KPAD, 0*32, col, half);
    wmma4(acc, gmemA(rs + 32, k0), sW1t, KPAD, 1*32, col, half);
    wmma4(acc, gmemA(rd,      k0), sW1t, KPAD, 2*32, col, half);
    wmma4(acc, gmemA(rd + 32, k0), sW1t, KPAD, 3*32, col, half);

    // K blocks 4..5: edge feats; stage f32 copy for residual
#pragma unroll
    for (int blk = 0; blk < 2; ++blk) {
      const int off = blk * 32;
      const float4* a0 = reinterpret_cast<const float4*>(re + off + k0);
      const float4* a1 = reinterpret_cast<const float4*>(re + off + 16 + k0);
      float4 f0 = a0[0], f1 = a0[1], f2 = a1[0], f3 = a1[1];
      float4* e0 = reinterpret_cast<float4*>(efstage + col*64 + off + k0);
      float4* e1 = reinterpret_cast<float4*>(efstage + col*64 + off + 16 + k0);
      e0[0] = f0; e0[1] = f1; e1[0] = f2; e1[1] = f3;
      wmma4(acc, cvt16(f0, f1, f2, f3), sW1t, KPAD, (4+blk)*32, col, half);
    }

    // K block 6: rbf d (16 valid, upper half zero)
    {
      const float4* q = reinterpret_cast<const float4*>(rr + k0);
      float4 f0 = q[0], f1 = q[1];
      v16bf a;
      a[0]=(bf)f0.x; a[1]=(bf)f0.y; a[2]=(bf)f0.z; a[3]=(bf)f0.w;
      a[4]=(bf)f1.x; a[5]=(bf)f1.y; a[6]=(bf)f1.z; a[7]=(bf)f1.w;
#pragma unroll
      for (int j = 8; j < 16; ++j) a[j] = (bf)0.f;
      wmma4(acc, a, sW1t, KPAD, 6*32, col, half);
    }

    // silu -> bf16 H tile in LDS (D layout -> row-major)
#pragma unroll
    for (int n = 0; n < 4; ++n)
#pragma unroll
      for (int r = 0; r < 8; ++r)
        hstage[(r + 8*half)*64 + col + 16*n] = (bf)silu(acc[n][r]);
    LDS_ORDER();   // LDS is in-order per wave; just stop compiler reordering

    // Layer 2: [16,64] @ [64,64]
    v8f acc2[4];
#pragma unroll
    for (int n = 0; n < 4; ++n) { v8f v;
#pragma unroll
      for (int r = 0; r < 8; ++r) v[r] = bias2[n]; acc2[n] = v; }
#pragma unroll
    for (int kb = 0; kb < 2; ++kb)
      wmma4(acc2, ldsA(hstage, col, 64, kb*32, k0), sW2t, 64, kb*32, col, half);

    // prefetch next tile's rows (index loads issued at loop top have long
    // completed; rows are 256B = two 128B lines)
    if (tn < EDGE_TILES) {
      const int en = tn*16 + col;
      const float* prs = ns + (size_t)is_n * 64;
      const float* prd = ns + (size_t)id_n * 64;
      __builtin_prefetch(prs,      0, 3); __builtin_prefetch(prs + 32, 0, 3);
      __builtin_prefetch(prd,      0, 3); __builtin_prefetch(prd + 32, 0, 3);
      const float* pre = ef + (size_t)en * 64;
      __builtin_prefetch(pre,      0, 3); __builtin_prefetch(pre + 32, 0, 3);
      __builtin_prefetch(dd + (size_t)en * 16, 0, 3);
    }

    // silu + residual + LayerNorm (rows live in 16-lane halves)
#pragma unroll
    for (int r = 0; r < 8; ++r) {
      float xv[4], s1 = 0.f, s2 = 0.f;
#pragma unroll
      for (int n = 0; n < 4; ++n) {
        float h = silu(acc2[n][r]);
        float e = efstage[(r + 8*half)*64 + col + 16*n];
        float x = e + h;
        xv[n] = x; s1 += x; s2 += x * x;
      }
#pragma unroll
      for (int m = 1; m < 16; m <<= 1) {
        s1 += __shfl_xor(s1, m, 32);
        s2 += __shfl_xor(s2, m, 32);
      }
      float mean = s1 * (1.f / 64.f);
      float var  = s2 * (1.f / 64.f) - mean * mean;
      float inv  = rsqrtf(var + 1e-5f);
      const size_t ob = (size_t)(ebase + r + 8*half) * 64;
#pragma unroll
      for (int n = 0; n < 4; ++n)
        out[ob + col + 16*n] = (xv[n] - mean) * inv * gam[n] + bet[n];
    }
  }
}

// ===================== Node position kernel =====================
__device__ __forceinline__ void gvp_vh(const float* vstage, float* vhstage, bf* astage,
                                       const float* Wh, int col, int half, int write_sh) {
  const int hb = half * 8;
  float vh[8][3];
#pragma unroll
  for (int h = 0; h < 8; ++h) { vh[h][0] = 0.f; vh[h][1] = 0.f; vh[h][2] = 0.f; }
  for (int vc = 0; vc < 16; ++vc) {
    float vx = vstage[col*48 + vc*3 + 0];
    float vy = vstage[col*48 + vc*3 + 1];
    float vz = vstage[col*48 + vc*3 + 2];
#pragma unroll
    for (int h = 0; h < 8; ++h) {
      float wv = Wh[vc*16 + hb + h];
      vh[h][0] = fmaf(vx, wv, vh[h][0]);
      vh[h][1] = fmaf(vy, wv, vh[h][1]);
      vh[h][2] = fmaf(vz, wv, vh[h][2]);
    }
  }
#pragma unroll
  for (int h = 0; h < 8; ++h) {
    vhstage[col*48 + (hb+h)*3 + 0] = vh[h][0];
    vhstage[col*48 + (hb+h)*3 + 1] = vh[h][1];
    vhstage[col*48 + (hb+h)*3 + 2] = vh[h][2];
    if (write_sh) {
      float nrm = sqrtf(vh[h][0]*vh[h][0] + vh[h][1]*vh[h][1] + vh[h][2]*vh[h][2] + 1e-8f);
      astage[col*96 + 64 + hb + h] = (bf)nrm;
    }
  }
}

__device__ __forceinline__ void gvp_vu_gate(const float* vhstage, float* vstage,
                                            const float* Wu, int col, int half) {
  const int ub = half * 8;
  float vu[8][3];
#pragma unroll
  for (int u = 0; u < 8; ++u) { vu[u][0] = 0.f; vu[u][1] = 0.f; vu[u][2] = 0.f; }
  for (int h = 0; h < 16; ++h) {
    float hx = vhstage[col*48 + h*3 + 0];
    float hy = vhstage[col*48 + h*3 + 1];
    float hz = vhstage[col*48 + h*3 + 2];
#pragma unroll
    for (int u = 0; u < 8; ++u) {
      float wv = Wu[h*16 + ub + u];
      vu[u][0] = fmaf(hx, wv, vu[u][0]);
      vu[u][1] = fmaf(hy, wv, vu[u][1]);
      vu[u][2] = fmaf(hz, wv, vu[u][2]);
    }
  }
#pragma unroll
  for (int u = 0; u < 8; ++u) {
    float vn = sqrtf(vu[u][0]*vu[u][0] + vu[u][1]*vu[u][1] + vu[u][2]*vu[u][2] + 1e-8f);
    float g = sigm(vn);
    vstage[col*48 + (ub+u)*3 + 0] = g * vu[u][0];
    vstage[col*48 + (ub+u)*3 + 1] = g * vu[u][1];
    vstage[col*48 + (ub+u)*3 + 2] = g * vu[u][2];
  }
}

__device__ __forceinline__ void gvp_gemm(bf* astage, const bf* Wst, const float* bsv,
                                         int col, int half, int k0) {
  v8f acc[4];
#pragma unroll
  for (int n = 0; n < 4; ++n) {
    float bv = bsv[col + 16*n];
    v8f v;
#pragma unroll
    for (int r = 0; r < 8; ++r) v[r] = bv;
    acc[n] = v;
  }
#pragma unroll
  for (int kb = 0; kb < 3; ++kb)
    wmma4(acc, ldsA(astage, col, 96, kb*32, k0), Wst, 96, kb*32, col, half);
  LDS_ORDER();
#pragma unroll
  for (int n = 0; n < 4; ++n)
#pragma unroll
    for (int r = 0; r < 8; ++r)
      astage[(r + 8*half)*96 + col + 16*n] = (bf)silu(acc[n][r]);
}

__global__ __launch_bounds__(128)
void node_pos_kernel(const float* __restrict__ ns, const float* __restrict__ pos,
                     const float* __restrict__ vec,
                     const float* __restrict__ Wh0, const float* __restrict__ Wu0,
                     const float* __restrict__ Ws0, const float* __restrict__ bs0,
                     const float* __restrict__ Wh1, const float* __restrict__ Wu1,
                     const float* __restrict__ Ws1, const float* __restrict__ bs1,
                     const float* __restrict__ Wh2, const float* __restrict__ Wu2,
                     float* __restrict__ outp) {
  __shared__ __align__(16) bf    sWst0[64 * NKPAD];   // 12288 B (K 80..95 zero)
  __shared__ __align__(16) bf    sWst1[64 * NKPAD];
  __shared__ __align__(16) float sWh0[256], sWu0[256], sWh1[256], sWu1[256], sWh2[256], sWu2[16];
  __shared__ __align__(16) bf    sA[4][16 * NKPAD];   // per-wave [s | sh | pad] bf16
  __shared__ __align__(16) float sV[4][16 * 48];      // per-wave v  (16 ch x 3)
  __shared__ __align__(16) float sVH[4][16 * 48];     // per-wave vh

  for (int i = threadIdx.x; i < 64 * NKPAD; i += blockDim.x) {
    int n = i / NKPAD, k = i % NKPAD;
    float w0 = (k < 80) ? Ws0[k*64 + n] : 0.f;
    float w1 = (k < 80) ? Ws1[k*64 + n] : 0.f;
    sWst0[i] = (bf)w0;
    sWst1[i] = (bf)w1;
  }
  for (int i = threadIdx.x; i < 256; i += blockDim.x) {
    sWh0[i] = Wh0[i]; sWu0[i] = Wu0[i];
    sWh1[i] = Wh1[i]; sWu1[i] = Wu1[i];
    sWh2[i] = Wh2[i];
  }
  if (threadIdx.x < 16) sWu2[threadIdx.x] = Wu2[threadIdx.x];
  __syncthreads();

  const int lane = threadIdx.x & 31;
  const int w    = threadIdx.x >> 5;
  const int col  = lane & 15;
  const int half = lane >> 4;
  const int k0   = half * 8;

  bf*    astage  = sA[w];
  float* vstage  = sV[w];
  float* vhstage = sVH[w];

  // zero K-pad cols 80..95 once (never overwritten afterwards)
  {
    v8bf z;
#pragma unroll
    for (int j = 0; j < 8; ++j) z[j] = (bf)0.f;
    *reinterpret_cast<v8bf*>(astage + col*96 + 80 + half*8) = z;
  }
  LDS_ORDER();

  for (int t = blockIdx.x * 4 + w; t < NODE_TILES; t += gridDim.x * 4) {
    const int nodebase = t * 16;
    const int g = nodebase + col;

    // stage s (bf16, cols half*32..+31) and v (f32)
    const float4* sp = reinterpret_cast<const float4*>(ns + (size_t)g*64 + half*32);
#pragma unroll
    for (int c8 = 0; c8 < 4; ++c8)
      *reinterpret_cast<v8bf*>(astage + col*96 + half*32 + c8*8) = cvt8(sp[2*c8], sp[2*c8+1]);
    const float4* vp = reinterpret_cast<const float4*>(vec + (size_t)g*48 + half*24);
    float4* vq = reinterpret_cast<float4*>(vstage + col*48 + half*24);
#pragma unroll
    for (int i = 0; i < 6; ++i) vq[i] = vp[i];
    LDS_ORDER();

    // prefetch next tile's node rows
    {
      const int tn = t + gridDim.x * 4;
      if (tn < NODE_TILES) {
        const int gn = tn * 16 + col;
        const float* pns = ns + (size_t)gn * 64;
        __builtin_prefetch(pns,      0, 3);
        __builtin_prefetch(pns + 32, 0, 3);
        __builtin_prefetch(vec + (size_t)gn * 48, 0, 3);
      }
    }

    // GVP 0
    gvp_vh(vstage, vhstage, astage, sWh0, col, half, 1);  LDS_ORDER();
    gvp_vu_gate(vhstage, vstage, sWu0, col, half);        LDS_ORDER();
    gvp_gemm(astage, sWst0, bs0, col, half, k0);          LDS_ORDER();
    // GVP 1
    gvp_vh(vstage, vhstage, astage, sWh1, col, half, 1);  LDS_ORDER();
    gvp_vu_gate(vhstage, vstage, sWu1, col, half);        LDS_ORDER();
    gvp_gemm(astage, sWst1, bs1, col, half, k0);          LDS_ORDER();
    // GVP 2: only the vector path feeds pos_out (scalar branch is dead)
    gvp_vh(vstage, vhstage, astage, sWh2, col, half, 0);  LDS_ORDER();

    float vux = 0.f, vuy = 0.f, vuz = 0.f;
#pragma unroll
    for (int h = 0; h < 16; ++h) {
      float wv = sWu2[h];
      vux = fmaf(vhstage[col*48 + h*3 + 0], wv, vux);
      vuy = fmaf(vhstage[col*48 + h*3 + 1], wv, vuy);
      vuz = fmaf(vhstage[col*48 + h*3 + 2], wv, vuz);
    }
    float vn = sqrtf(vux*vux + vuy*vuy + vuz*vuz + 1e-8f);
    if (half == 0) {
      outp[(size_t)g*3 + 0] = pos[(size_t)g*3 + 0] + vn * vux;
      outp[(size_t)g*3 + 1] = pos[(size_t)g*3 + 1] + vn * vuy;
      outp[(size_t)g*3 + 2] = pos[(size_t)g*3 + 2] + vn * vuz;
    }
  }
}

extern "C" void kernel_launch(void* const* d_in, const int* in_sizes, int n_in,
                              void* d_out, int out_size, void* d_ws, size_t ws_size,
                              hipStream_t stream) {
  const float* ns   = (const float*)d_in[0];
  const float* pos  = (const float*)d_in[1];
  const float* vec  = (const float*)d_in[2];
  const float* ef   = (const float*)d_in[3];
  const float* dd   = (const float*)d_in[4];
  const int*   src  = (const int*)d_in[5];
  const int*   dst  = (const int*)d_in[6];
  const float* W1   = (const float*)d_in[7];
  const float* b1   = (const float*)d_in[8];
  const float* W2   = (const float*)d_in[9];
  const float* b2   = (const float*)d_in[10];
  const float* lng  = (const float*)d_in[11];
  const float* lnb  = (const float*)d_in[12];
  const float* Wh0  = (const float*)d_in[13];
  const float* Wu0  = (const float*)d_in[14];
  const float* Ws0  = (const float*)d_in[15];
  const float* bs0  = (const float*)d_in[16];
  const float* Wh1  = (const float*)d_in[17];
  const float* Wu1  = (const float*)d_in[18];
  const float* Ws1  = (const float*)d_in[19];
  const float* bs1  = (const float*)d_in[20];
  const float* Wh2  = (const float*)d_in[21];
  const float* Wu2  = (const float*)d_in[22];
  // d_in[23], d_in[24] (gvp2_Ws, gvp2_bs) feed a dead scalar branch

  float* out      = (float*)d_out;
  float* out_pos  = out + (size_t)E_TOTAL * FDIM;

  edge_update_kernel<<<dim3(1024), dim3(128), 0, stream>>>(
      ns, ef, dd, src, dst, W1, b1, W2, b2, lng, lnb, out);
  node_pos_kernel<<<dim3(512), dim3(128), 0, stream>>>(
      ns, pos, vec, Wh0, Wu0, Ws0, bs0, Wh1, Wu1, Ws1, bs1, Wh2, Wu2, out_pos);
}
